// DotGeneralModule_27066883899824
// MI455X (gfx1250) — compile-verified, matrix operand order checked by reading, not run on hardware
//
#include <hip/hip_runtime.h>
#include <stdint.h>

typedef __attribute__((ext_vector_type(8))) int v8i;
typedef __attribute__((ext_vector_type(2))) int v2i;

#define MDIM 4096
#define NDIM 4096
#define KDIM 4096

// ---------------------------------------------------------------------------
// Stage 1: per-tensor abs-max reduction (float bits compare == float compare
// for non-negative values, so atomicMax on the bit pattern is exact).
// ---------------------------------------------------------------------------
__global__ void absmax_kernel(const float* __restrict__ x,
                              unsigned int* __restrict__ amax) {
  const float4* x4 = (const float4*)x;
  const size_t n4 = (size_t)MDIM * KDIM / 4;
  const size_t stride = (size_t)gridDim.x * blockDim.x;
  float m = 0.0f;
  for (size_t j = (size_t)blockIdx.x * blockDim.x + threadIdx.x; j < n4; j += stride) {
    float4 v = x4[j];
    m = fmaxf(m, fmaxf(fmaxf(fabsf(v.x), fabsf(v.y)),
                       fmaxf(fabsf(v.z), fabsf(v.w))));
  }
#pragma unroll
  for (int off = 16; off > 0; off >>= 1)
    m = fmaxf(m, __shfl_down(m, off, 32));

  __shared__ float smax[8];
  const int lane = threadIdx.x & 31;
  const int wid  = threadIdx.x >> 5;
  if (lane == 0) smax[wid] = m;
  __syncthreads();
  if (wid == 0) {
    m = (lane < 8) ? smax[lane] : 0.0f;
#pragma unroll
    for (int off = 4; off > 0; off >>= 1)
      m = fmaxf(m, __shfl_down(m, off, 32));
    if (lane == 0) atomicMax(amax, __float_as_uint(m));
  }
}

// ---------------------------------------------------------------------------
// Stage 2: scales. sc[2]=ls, sc[3]=rs, sc[4]=1/(ls*rs)
// ---------------------------------------------------------------------------
__global__ void scales_kernel(const unsigned int* __restrict__ amax,
                              float* __restrict__ sc) {
  if (threadIdx.x == 0) {
    float al = fmaxf(__uint_as_float(amax[0]), 1e-12f);
    float ar = fmaxf(__uint_as_float(amax[1]), 1e-12f);
    float ls = 127.0f / al;
    float rs = 127.0f / ar;
    sc[2] = ls;
    sc[3] = rs;
    sc[4] = 1.0f / (ls * rs);
  }
}

__device__ __forceinline__ int quant1(float v, float s) {
  int q = __float2int_rn(v * s);         // round-to-nearest-even, like jnp.round
  return min(127, max(-127, q));
}

// ---------------------------------------------------------------------------
// Stage 3: quantize lhs, row-major M x K, 4 elements -> 1 packed dword.
// ---------------------------------------------------------------------------
__global__ void quantA_kernel(const float* __restrict__ x,
                              int8_t* __restrict__ q,
                              const float* __restrict__ sp) {
  const float s = sp[0];
  const size_t n4 = (size_t)MDIM * KDIM / 4;
  size_t j = (size_t)blockIdx.x * blockDim.x + threadIdx.x;
  if (j >= n4) return;
  float4 v = ((const float4*)x)[j];
  unsigned int pack = ((unsigned)(quant1(v.x, s) & 0xFF))
                    | ((unsigned)(quant1(v.y, s) & 0xFF) << 8)
                    | ((unsigned)(quant1(v.z, s) & 0xFF) << 16)
                    | ((unsigned)(quant1(v.w, s) & 0xFF) << 24);
  ((unsigned int*)q)[j] = pack;
}

// ---------------------------------------------------------------------------
// Stage 4: quantize rhs AND transpose: input K x N row-major -> output N x K
// (B^T row-major) so WMMA B-fragment loads are contiguous 8-byte chunks.
// Block (32,8), 32x32 tile via LDS.
// ---------------------------------------------------------------------------
__global__ void quantB_kernel(const float* __restrict__ x,
                              int8_t* __restrict__ q,
                              const float* __restrict__ sp) {
  __shared__ int8_t tile[32][33];
  const float s = sp[0];
  const int n0 = blockIdx.x * 32;
  const int k0 = blockIdx.y * 32;
#pragma unroll
  for (int r = 0; r < 32; r += 8) {
    int k = k0 + threadIdx.y + r;
    int n = n0 + threadIdx.x;
    float v = x[(size_t)k * NDIM + n];
    tile[threadIdx.y + r][threadIdx.x] = (int8_t)quant1(v, s);
  }
  __syncthreads();
  const int tid = threadIdx.y * 32 + threadIdx.x;
  const int nl = tid >> 3;        // 0..31 output row (n)
  const int kl = (tid & 7) * 4;   // 0..28 output col (k), 4 bytes each
  unsigned int pack = ((unsigned)(tile[kl + 0][nl] & 0xFF))
                    | ((unsigned)(tile[kl + 1][nl] & 0xFF) << 8)
                    | ((unsigned)(tile[kl + 2][nl] & 0xFF) << 16)
                    | ((unsigned)(tile[kl + 3][nl] & 0xFF) << 24);
  *(unsigned int*)(q + (size_t)(n0 + nl) * KDIM + k0 + kl) = pack;
}

// ---------------------------------------------------------------------------
// Stage 5: int8 WMMA GEMM. qA: M x K row-major. qB: N x K row-major (B^T).
// 8 waves / block (2 M-waves x 4 N-waves); wave tile 32x64 = 2x4 of 16x16.
// Block tile 64x256. Per K-step of 64: 6 fragment loads (6 KB) feed 8 WMMAs
// (262 Kops) -> ~43 int-ops per L1 byte; A frags reused 4x, B frags 2x.
// Per-lane fragment layout (8-bit A/B, wave32):
//   lane<16 : row/col = lane,     K chunks {0-7,16-23,32-39,48-55}
//   lane>=16: row/col = lane-16,  K chunks {8-15,24-31,40-47,56-63}
// ---------------------------------------------------------------------------
__global__ void __launch_bounds__(256, 2)
int8_wmma_gemm(const int8_t* __restrict__ qA, const int8_t* __restrict__ qB,
               float* __restrict__ out, const float* __restrict__ sc) {
  const float inv = sc[4];
  const int lane = threadIdx.x & 31;
  const int wave = threadIdx.x >> 5;
  const int waveM = wave & 1;   // 0..1 -> 32 rows each
  const int waveN = wave >> 1;  // 0..3 -> 64 cols each
  const int l16 = lane & 15;
  const int half = lane >> 4;

  const int tileM = blockIdx.y * 64 + waveM * 32;
  const int tileN = blockIdx.x * 256 + waveN * 64;

  const int8_t* aRow[2];
  const int8_t* bRow[4];
#pragma unroll
  for (int i = 0; i < 2; ++i)
    aRow[i] = qA + (size_t)(tileM + i * 16 + l16) * KDIM + half * 8;
#pragma unroll
  for (int i = 0; i < 4; ++i)
    bRow[i] = qB + (size_t)(tileN + i * 16 + l16) * KDIM + half * 8;

  v8i acc[2][4] = {};

  for (int k0 = 0; k0 < KDIM; k0 += 64) {
    v8i a[2], b[4];
#pragma unroll
    for (int i = 0; i < 2; ++i) {
#pragma unroll
      for (int c = 0; c < 4; ++c) {
        v2i t = *(const v2i*)(aRow[i] + k0 + c * 16);
        a[i][2 * c] = t[0];
        a[i][2 * c + 1] = t[1];
      }
    }
#pragma unroll
    for (int i = 0; i < 4; ++i) {
#pragma unroll
      for (int c = 0; c < 4; ++c) {
        v2i t = *(const v2i*)(bRow[i] + k0 + c * 16);
        b[i][2 * c] = t[0];
        b[i][2 * c + 1] = t[1];
      }
    }
#pragma unroll
    for (int mi = 0; mi < 2; ++mi)
#pragma unroll
      for (int ni = 0; ni < 4; ++ni)
        acc[mi][ni] = __builtin_amdgcn_wmma_i32_16x16x64_iu8(
            /*sgn_a=*/true, a[mi], /*sgn_b=*/true, b[ni], acc[mi][ni],
            /*reuse_a=*/false, /*reuse_b=*/false);
  }

  // C/D layout: VGPR r -> (lanes 0-15: M=r, N=lane) (lanes 16-31: M=8+r, N=lane-16)
#pragma unroll
  for (int mi = 0; mi < 2; ++mi) {
#pragma unroll
    for (int ni = 0; ni < 4; ++ni) {
      const int col = tileN + ni * 16 + l16;
      const int rowbase = tileM + mi * 16 + half * 8;
#pragma unroll
      for (int r = 0; r < 8; ++r) {
        out[(size_t)(rowbase + r) * NDIM + col] = (float)acc[mi][ni][r] * inv;
      }
    }
  }
}

// ---------------------------------------------------------------------------
extern "C" void kernel_launch(void* const* d_in, const int* in_sizes, int n_in,
                              void* d_out, int out_size, void* d_ws, size_t ws_size,
                              hipStream_t stream) {
  const float* lhs = (const float*)d_in[0];
  const float* rhs = (const float*)d_in[1];
  float* out = (float*)d_out;

  uint8_t* ws = (uint8_t*)d_ws;
  unsigned int* amax = (unsigned int*)ws;     // [0]=lhs amax bits, [1]=rhs
  float* scales = (float*)ws;                 // [2]=ls, [3]=rs, [4]=1/(ls*rs)
  int8_t* qA = (int8_t*)(ws + 256);
  int8_t* qB = qA + (size_t)MDIM * KDIM;

  // zero the atomicMax cells (graph-capture-safe)
  hipMemsetAsync(d_ws, 0, 2 * sizeof(unsigned int), stream);

  absmax_kernel<<<2048, 256, 0, stream>>>(lhs, amax + 0);
  absmax_kernel<<<2048, 256, 0, stream>>>(rhs, amax + 1);
  scales_kernel<<<1, 32, 0, stream>>>(amax, scales);

  const size_t n4 = (size_t)MDIM * KDIM / 4;
  quantA_kernel<<<(unsigned)((n4 + 255) / 256), 256, 0, stream>>>(lhs, qA, scales + 2);

  dim3 tb(32, 8);
  dim3 tg(NDIM / 32, KDIM / 32);
  quantB_kernel<<<tg, tb, 0, stream>>>(rhs, qB, scales + 3);

  dim3 gg(NDIM / 256, MDIM / 64);
  int8_wmma_gemm<<<gg, 256, 0, stream>>>(qA, qB, out, scales);
}
